// RBFNet_23167053595193
// MI455X (gfx1250) — matrix operand order repaired and checked by reading, not compiled, not run on hardware
//
#include <hip/hip_runtime.h>
#include <stdint.h>

#define B_ROWS 16384
#define D_DIM  512
#define H_DIM  1024
#define M_TILE 64             // B-rows per workgroup
#define N_STAGE 32            // centers rows per staged tile (2 waves x 16)
#define NT (H_DIM / N_STAGE)  // 32 tiles
#define CSTR 520              // bf16 LDS row stride: 1040B == 16 mod 256 -> conflict-free frags

typedef float  v4f  __attribute__((ext_vector_type(4)));
typedef float  v8f  __attribute__((ext_vector_type(8)));
typedef __bf16 v4bf __attribute__((ext_vector_type(4)));
typedef __bf16 v8bf __attribute__((ext_vector_type(8)));
typedef __bf16 v16bf __attribute__((ext_vector_type(16)));

__device__ __forceinline__ void async_ld_b128(uint32_t lds_byte_off,
                                              const float* sbase,
                                              uint32_t g_byte_off) {
  // GVS mode: LDS[vdst] <= MEM[saddr64 + vaddr32], tracked by ASYNCcnt
  asm volatile("global_load_async_to_lds_b128 %0, %1, %2"
               :: "v"(lds_byte_off), "v"(g_byte_off), "s"(sbase)
               : "memory");
}

__device__ __forceinline__ v16bf cat16(v8bf lo, v8bf hi) {
  return __builtin_shufflevector(lo, hi, 0,1,2,3,4,5,6,7,8,9,10,11,12,13,14,15);
}

#define WMMA_BF16(A, Bm, Cm) \
  __builtin_amdgcn_wmma_f32_16x16x32_bf16(false, (A), false, (Bm), (short)0, (Cm), false, false)

// Precompute csq[h] = sum_d centers[h][d]^2 into workspace.
__global__ __launch_bounds__(256, 1)
void rbf_csq_kernel(const float* __restrict__ centers, float* __restrict__ csq) {
  const int lane = threadIdx.x & 31;
  const int row  = blockIdx.x * 8 + (threadIdx.x >> 5);
  const float* p = centers + row * D_DIM;
  float s = 0.f;
  #pragma unroll
  for (int i = 0; i < D_DIM / 32; ++i) {
    float v = p[lane + i * 32];
    s += v * v;
  }
  #pragma unroll
  for (int off = 16; off > 0; off >>= 1) s += __shfl_xor(s, off, 32);
  if (lane == 0) csq[row] = s;
}

__global__ __launch_bounds__(256, 1)
void rbf_main_kernel(const float* __restrict__ x,
                     const float* __restrict__ centers,
                     const float* __restrict__ sigmas,
                     const float* __restrict__ W,
                     const float* __restrict__ bvec,
                     const float* __restrict__ csq,
                     float* __restrict__ out) {
  __shared__ float  sbuf[N_STAGE * D_DIM];   // 64 KB f32 async-staging buffer
  __shared__ __bf16 xh[M_TILE * CSTR];       // 65 KB  x hi
  __shared__ __bf16 xl[M_TILE * CSTR];       // 65 KB  x lo
  __shared__ __bf16 ch[N_STAGE * CSTR];      // 32.5 KB centers hi
  __shared__ __bf16 cl[N_STAGE * CSTR];      // 32.5 KB centers lo
  __shared__ float  xsq[M_TILE];
  __shared__ float  partial[8][16];

  const int tid  = threadIdx.x;
  const int lane = tid & 31;
  const int wave = tid >> 5;
  const int rg   = wave & 3;            // M row-group (16 rows)
  const int hs   = wave >> 2;           // which 16-row half of the staged 32 H rows
  const int nn   = lane & 15;
  const int g8   = (lane >> 4) << 3;    // A-frag K sub-offset: 0 or 8
  const int bg   = (lane >> 4) << 4;    // unused; B frags use g8, same k order as A
  const int row0 = blockIdx.x * M_TILE;

  const uint32_t sb = (uint32_t)(uintptr_t)&sbuf[0];

  if (tid < M_TILE) xsq[tid] = 0.f;
  __syncthreads();

  // ---- stage + split-convert x (two 32-row rounds through sbuf) ----
  for (int rd = 0; rd < 2; ++rd) {
    #pragma unroll
    for (int i = 0; i < 16; ++i) {
      int j = i * 256 + tid;
      int r = j >> 7;            // 0..31
      int c4 = j & 127;
      async_ld_b128(sb + (uint32_t)((r * D_DIM + c4 * 4) * 4),
                    x, (uint32_t)(((row0 + rd * 32 + r) * D_DIM + c4 * 4) * 4));
    }
    asm volatile("s_wait_asynccnt 0" ::: "memory");
    __syncthreads();
    #pragma unroll 4
    for (int i = 0; i < 16; ++i) {
      int j = i * 256 + tid;
      int r = j >> 7;
      int c4 = j & 127;
      v4f v = *(const v4f*)&sbuf[r * D_DIM + c4 * 4];
      v4bf h, l;
      float s = 0.f;
      #pragma unroll
      for (int e = 0; e < 4; ++e) {
        float f = v[e];
        __bf16 hb = (__bf16)f;
        h[e] = hb;
        l[e] = (__bf16)(f - (float)hb);
        s += f * f;
      }
      int gr = rd * 32 + r;
      *(v4bf*)&xh[gr * CSTR + c4 * 4] = h;
      *(v4bf*)&xl[gr * CSTR + c4 * 4] = l;
      atomicAdd(&xsq[gr], s);            // ds_add_f32
    }
    __syncthreads();
  }

  // Hoist this lane's 8 row norms (rows rg*16 + r + 8*(lane>=16)).
  float xr[8];
  #pragma unroll
  for (int r = 0; r < 8; ++r) xr[r] = xsq[rg * 16 + r + g8];  // g8 == 8*(lane>=16)

  // ---- prefetch centers tile 0 ----
  #pragma unroll
  for (int i = 0; i < 16; ++i) {
    int j = i * 256 + tid;
    int r = j >> 7;
    int c4 = j & 127;
    async_ld_b128(sb + (uint32_t)((r * D_DIM + c4 * 4) * 4),
                  centers, (uint32_t)((r * D_DIM + c4 * 4) * 4));
  }

  float acc[8] = {0.f,0.f,0.f,0.f,0.f,0.f,0.f,0.f};
  const __bf16* arow_h = &xh[(rg * 16 + nn) * CSTR];
  const __bf16* arow_l = &xl[(rg * 16 + nn) * CSTR];
  const __bf16* brow_h = &ch[(hs * 16 + nn) * CSTR];
  const __bf16* brow_l = &cl[(hs * 16 + nn) * CSTR];

  for (int it = 0; it < NT; ++it) {
    asm volatile("s_wait_asynccnt 0" ::: "memory");   // tile `it` f32 landed
    __syncthreads();                                  // ...and prior WMMA done with ch/cl

    // convert sbuf (f32) -> ch/cl (bf16 hi/lo)
    #pragma unroll 4
    for (int i = 0; i < 16; ++i) {
      int j = i * 256 + tid;
      int r = j >> 7;
      int c4 = j & 127;
      v4f v = *(const v4f*)&sbuf[r * D_DIM + c4 * 4];
      v4bf h, l;
      #pragma unroll
      for (int e = 0; e < 4; ++e) {
        float f = v[e];
        __bf16 hb = (__bf16)f;
        h[e] = hb;
        l[e] = (__bf16)(f - (float)hb);
      }
      *(v4bf*)&ch[r * CSTR + c4 * 4] = h;
      *(v4bf*)&cl[r * CSTR + c4 * 4] = l;
    }
    __syncthreads();   // ch/cl ready; sbuf free for refill

    if (it + 1 < NT) { // overlap next tile's async DMA with WMMA + epilogue
      const int hb2 = (it + 1) * N_STAGE;
      #pragma unroll
      for (int i = 0; i < 16; ++i) {
        int j = i * 256 + tid;
        int r = j >> 7;
        int c4 = j & 127;
        async_ld_b128(sb + (uint32_t)((r * D_DIM + c4 * 4) * 4),
                      centers, (uint32_t)(((hb2 + r) * D_DIM + c4 * 4) * 4));
      }
    }

    // ---- 16 K-chunks x 3 bf16 WMMAs (hh, hl, lh) ----
    v8f d0 = {}; v8f d1 = {}; v8f d2 = {}; v8f d3 = {}; v8f d4 = {}; v8f d5 = {};
    #pragma unroll 4
    for (int c = 0; c < 16; ++c) {
      const int k = c * 32;
      v16bf Ah = cat16(*(const v8bf*)(arow_h + k + g8),
                       *(const v8bf*)(arow_h + k + 16 + g8));
      v16bf Al = cat16(*(const v8bf*)(arow_l + k + g8),
                       *(const v8bf*)(arow_l + k + 16 + g8));
      v16bf Bh = cat16(*(const v8bf*)(brow_h + k + g8),
                       *(const v8bf*)(brow_h + k + 16 + g8));
      v16bf Bl = cat16(*(const v8bf*)(brow_l + k + g8),
                       *(const v8bf*)(brow_l + k + 16 + g8));
      if (c & 1) {
        d3 = WMMA_BF16(Ah, Bh, d3);
        d4 = WMMA_BF16(Ah, Bl, d4);
        d5 = WMMA_BF16(Al, Bh, d5);
      } else {
        d0 = WMMA_BF16(Ah, Bh, d0);
        d1 = WMMA_BF16(Ah, Bl, d1);
        d2 = WMMA_BF16(Al, Bh, d2);
      }
    }
    v8f C = ((d0 + d3) + (d1 + d4)) + (d2 + d5);   // C[r] = x.c

    const int h = it * N_STAGE + hs * 16 + nn;
    const float cs   = csq[h];
    const float sg   = sigmas[h];
    const float inv2 = 1.0f / (2.f * sg * sg + 1e-6f);
    const float w    = W[h];
    #pragma unroll
    for (int r = 0; r < 8; ++r) {
      float dist = xr[r] + cs - 2.0f * C[r];
      dist = fmaxf(dist, 0.0f);
      acc[r] += __expf(-dist * inv2) * w;
    }
  }

  // Reduce over the 16 N-lanes of each half.
  #pragma unroll
  for (int r = 0; r < 8; ++r) {
    acc[r] += __shfl_xor(acc[r], 1, 32);
    acc[r] += __shfl_xor(acc[r], 2, 32);
    acc[r] += __shfl_xor(acc[r], 4, 32);
    acc[r] += __shfl_xor(acc[r], 8, 32);
  }
  if (nn == 0) {                 // lanes 0 (rows 0-7) and 16 (rows 8-15)
    const int half = lane >> 4;
    #pragma unroll
    for (int r = 0; r < 8; ++r) partial[wave][half * 8 + r] = acc[r];
  }
  __syncthreads();
  if (tid < M_TILE) {            // combine the two H-split waves per row group
    const int rg2 = tid >> 4;
    const int rr  = tid & 15;
    float v = partial[rg2][rr] + partial[rg2 + 4][rr] + bvec[0];
    if (v != v) v = 0.f;         // NaN guard
    out[row0 + tid] = v;
  }
}

extern "C" void kernel_launch(void* const* d_in, const int* in_sizes, int n_in,
                              void* d_out, int out_size, void* d_ws, size_t ws_size,
                              hipStream_t stream) {
  (void)in_sizes; (void)n_in; (void)out_size; (void)ws_size;
  const float* x       = (const float*)d_in[0];
  const float* centers = (const float*)d_in[1];
  const float* sigmas  = (const float*)d_in[2];
  const float* W       = (const float*)d_in[3];
  const float* b       = (const float*)d_in[4];
  float* out = (float*)d_out;
  float* csq = (float*)d_ws;   // H_DIM floats = 4 KB

  rbf_csq_kernel<<<H_DIM / 8, 256, 0, stream>>>(centers, csq);
  rbf_main_kernel<<<B_ROWS / M_TILE, 256, 0, stream>>>(x, centers, sigmas, W, b, csq, out);
}
